// MHA_fused_86122684219586
// MI455X (gfx1250) — matrix operand order repaired and checked by reading, not compiled, not run on hardware
//
#include <hip/hip_runtime.h>
#include <stdint.h>

typedef __attribute__((ext_vector_type(16))) _Float16 v16h;
typedef __attribute__((ext_vector_type(8)))  _Float16 v8h;
typedef __attribute__((ext_vector_type(8)))  float    v8f;

#define B_   2
#define S_   2048
#define DM   1024
#define H_   16
#define HD   64
#define NCOL 3072
#define ROWS (B_*S_)
#define SCALE_F 0.125f            /* 64^-0.5 */
#define MASK_FILL_F (-1000000000.0f)

// ---------------------------------------------------------------------------
// CDNA5 async copy: global -> LDS, 16 bytes per lane, tracked by ASYNCcnt.
// lds_off: per-lane LDS byte address (low 32 bits of a generic shared ptr).
// ---------------------------------------------------------------------------
__device__ __forceinline__ void async_copy_b128(uint32_t lds_off, const void* gptr) {
  asm volatile("global_load_async_to_lds_b128 %0, %1, off"
               :: "v"(lds_off), "v"(gptr) : "memory");
}
__device__ __forceinline__ void wait_async0() {
  asm volatile("s_wait_asynccnt 0x0" ::: "memory");
}
__device__ __forceinline__ uint32_t lds_off_of(const void* p) {
  return (uint32_t)(uintptr_t)p;
}

// ---------------------------------------------------------------------------
// Kernel 1: fp32 -> f16 conversion
// ---------------------------------------------------------------------------
__global__ void cvt_f32_f16(const float* __restrict__ src,
                            _Float16* __restrict__ dst, int n) {
  int i = blockIdx.x * blockDim.x + threadIdx.x;
  if (i < n) dst[i] = (_Float16)src[i];
}

// ---------------------------------------------------------------------------
// Kernel 2: QKV projection. C[r,n] = sum_m X[r,m] * W[n,m]
// Block: 128 threads (4 waves). Tile: 64 rows x 64 cols. K staged 32 at a time
// via global_load_async_to_lds_b128 (256 chunks of 16B per operand tile).
// ---------------------------------------------------------------------------
__global__ __launch_bounds__(128)
void qkv_gemm(const _Float16* __restrict__ Xh, const _Float16* __restrict__ Wh,
              _Float16* __restrict__ Qh, _Float16* __restrict__ Kh,
              _Float16* __restrict__ Vh) {
  __shared__ _Float16 As[64][32];   // [row][k]
  __shared__ _Float16 Bs[64][32];   // [col][k]

  const int tid   = threadIdx.x;
  const int wv    = tid >> 5;
  const int lane  = tid & 31;
  const int lhalf = lane >> 4;      // 0 | 1
  const int l16   = lane & 15;
  const int row0  = blockIdx.y * 64;
  const int col0  = blockIdx.x * 64;

  const uint32_t asBase = lds_off_of(&As[0][0]);
  const uint32_t bsBase = lds_off_of(&Bs[0][0]);

  const v8f vzero = {};
  v8f acc[4];
#pragma unroll
  for (int t = 0; t < 4; ++t) acc[t] = vzero;

  for (int k0 = 0; k0 < DM; k0 += 32) {
    __syncthreads();          // previous iteration's LDS reads complete
    // ---- async stage A (64x32 f16) and B (64x32 f16): 2 chunks each/lane ----
#pragma unroll
    for (int p = 0; p < 2; ++p) {
      int c  = tid + p * 128;       // chunk id in [0,256)
      int r  = c >> 2;              // 4 x 16B chunks per 32-f16 row
      int c8 = (c & 3) * 8;
      async_copy_b128(asBase + c * 16, Xh + (size_t)(row0 + r) * DM + k0 + c8);
      async_copy_b128(bsBase + c * 16, Wh + (size_t)(col0 + r) * DM + k0 + c8);
    }
    if (k0 + 32 < DM) {   // global_prefetch_b8 of next K slab
      __builtin_prefetch(Xh + (size_t)(row0 + (tid & 63)) * DM + (k0 + 32), 0, 1);
      __builtin_prefetch(Wh + (size_t)(col0 + (tid & 63)) * DM + (k0 + 32), 0, 1);
    }
    wait_async0();
    __syncthreads();

    // A fragment: 16x32 f16, lane = M (mod 16), lane-half selects K groups.
    v16h a;
#pragma unroll
    for (int i = 0; i < 8; ++i) {
      int kk = (i < 4) ? (lhalf * 8 + 2 * i) : (16 + lhalf * 8 + 2 * (i - 4));
      a[2 * i]     = As[wv * 16 + l16][kk];
      a[2 * i + 1] = As[wv * 16 + l16][kk + 1];
    }
#pragma unroll
    for (int t = 0; t < 4; ++t) {
      // B fragment: 32x16 f16, lane = N (mod 16), lane-half selects K half.
      v16h bf;
#pragma unroll
      for (int i = 0; i < 8; ++i) {
        int kk = (i < 4) ? (lhalf * 8 + 2 * i) : (16 + lhalf * 8 + 2 * (i - 4));
        bf[2 * i]     = Bs[t * 16 + l16][kk];
        bf[2 * i + 1] = Bs[t * 16 + l16][kk + 1];
      }
      acc[t] = __builtin_amdgcn_wmma_f32_16x16x32_f16(
          false, a, false, bf, (short)0, acc[t], false, false);
    }
  }

  // Scatter C (16x16 f32: VGPR i -> M = i + lhalf*8, lane -> N = l16)
#pragma unroll
  for (int t = 0; t < 4; ++t) {
#pragma unroll
    for (int i = 0; i < 8; ++i) {
      int r = row0 + wv * 16 + i + lhalf * 8;       // global row (b*S + s)
      int n = col0 + t * 16 + l16;                  // global col in [0,3072)
      int b = r >> 11, s = r & (S_ - 1);
      int c   = n >> 10;
      int rem = n & 1023;
      int h   = rem >> 6;
      int d   = rem & 63;
      _Float16* dst = (c == 0) ? Qh : (c == 1) ? Kh : Vh;
      dst[(((size_t)(b * H_ + h)) * S_ + s) * HD + d] = (_Float16)acc[t][i];
    }
  }
}

// ---------------------------------------------------------------------------
// Kernel 3: flash attention. Block: 128 threads (4 waves) = 64 queries of one
// (b,h). 32 keys per iteration; online softmax; WMMA for QK^T and PV.
// Q and K tiles staged with async-to-LDS; V staged transposed (b128 reg loads
// + ds_store_b16 scatter).
// ---------------------------------------------------------------------------
__global__ __launch_bounds__(128)
void flash_attn(const _Float16* __restrict__ Qh, const _Float16* __restrict__ Kh,
                const _Float16* __restrict__ Vh, const uint8_t* __restrict__ mask,
                float* __restrict__ out) {
  __shared__ _Float16 Qs[64][64];      // [q][d]
  __shared__ _Float16 Ks[32][64];      // [key][d]
  __shared__ _Float16 Vt[64][32];      // [d][key]  (transposed for B-frag pairs)
  __shared__ _Float16 Ps[4][16][32];   // per-wave P tile [q][key]

  const int tid   = threadIdx.x;
  const int wv    = tid >> 5;
  const int lane  = tid & 31;
  const int lhalf = lane >> 4;
  const int l16   = lane & 15;
  const int q0    = blockIdx.x * 64;
  const int bh    = blockIdx.y;
  const int b     = bh >> 4;
  const int h     = bh & 15;
  const size_t headBase = ((size_t)(b * H_ + h)) * S_ * HD;

  const uint32_t qsBase = lds_off_of(&Qs[0][0]);
  const uint32_t ksBase = lds_off_of(&Ks[0][0]);

  // ---- async stage Q tile: 64x64 f16 = 512 x 16B chunks, 4 per lane ----
#pragma unroll
  for (int p = 0; p < 4; ++p) {
    int c  = tid + p * 128;
    int r  = c >> 3;                 // 8 x 16B chunks per 64-f16 row
    int c8 = (c & 7) * 8;
    async_copy_b128(qsBase + c * 16, Qh + headBase + (size_t)(q0 + r) * HD + c8);
  }
  wait_async0();
  __syncthreads();

  // Q A-fragments for the two d-halves (K-dim of QK^T)
  v16h qa[2];
#pragma unroll
  for (int j = 0; j < 2; ++j) {
#pragma unroll
    for (int i = 0; i < 8; ++i) {
      int kk = (i < 4) ? (lhalf * 8 + 2 * i) : (16 + lhalf * 8 + 2 * (i - 4));
      qa[j][2 * i]     = Qs[wv * 16 + l16][j * 32 + kk];
      qa[j][2 * i + 1] = Qs[wv * 16 + l16][j * 32 + kk + 1];
    }
  }

  const v8f vzero = {};
  float m_i[8], l_i[8];
  v8f o[4];
#pragma unroll
  for (int i = 0; i < 8; ++i) { m_i[i] = -1e30f; l_i[i] = 0.0f; }
#pragma unroll
  for (int t = 0; t < 4; ++t) o[t] = vzero;

  for (int kb = 0; kb < S_ / 32; ++kb) {
    __syncthreads();
    // ---- async stage K tile: 32x64 f16 = 256 chunks, 2 per lane ----
#pragma unroll
    for (int p = 0; p < 2; ++p) {
      int c  = tid + p * 128;
      int kk = c >> 3;
      int c8 = (c & 7) * 8;
      async_copy_b128(ksBase + c * 16,
                      Kh + headBase + (size_t)(kb * 32 + kk) * HD + c8);
    }
    // ---- stage V transposed: b128 register load + ds_store_b16 scatter ----
#pragma unroll
    for (int p = 0; p < 2; ++p) {
      int c  = tid + p * 128;
      int kk = c >> 3;
      int d8 = (c & 7) * 8;
      v8h vv = *(const v8h*)(Vh + headBase + (size_t)(kb * 32 + kk) * HD + d8);
#pragma unroll
      for (int j = 0; j < 8; ++j) Vt[d8 + j][kk] = vv[j];
    }
    wait_async0();
    __syncthreads();

    // ---- logits: S = Q * K^T * scale, masked ----
    float sv[2][8];
#pragma unroll
    for (int kt = 0; kt < 2; ++kt) {
      v16h b0, b1;   // B-frags: N = key (kt*16 + l16), K = d
#pragma unroll
      for (int i = 0; i < 8; ++i) {
        int dd = (i < 4) ? (lhalf * 8 + 2 * i) : (16 + lhalf * 8 + 2 * (i - 4));
        b0[2 * i]     = Ks[kt * 16 + l16][dd];
        b0[2 * i + 1] = Ks[kt * 16 + l16][dd + 1];
        b1[2 * i]     = Ks[kt * 16 + l16][32 + dd];
        b1[2 * i + 1] = Ks[kt * 16 + l16][32 + dd + 1];
      }
      v8f c = vzero;
      c = __builtin_amdgcn_wmma_f32_16x16x32_f16(false, qa[0], false, b0,
                                                 (short)0, c, false, false);
      c = __builtin_amdgcn_wmma_f32_16x16x32_f16(false, qa[1], false, b1,
                                                 (short)0, c, false, false);
#pragma unroll
      for (int i = 0; i < 8; ++i) {
        int qidx = q0 + wv * 16 + i + lhalf * 8;
        int key  = kb * 32 + kt * 16 + l16;
        float s  = c[i] * SCALE_F;
        uint8_t mb = mask[(((size_t)(b * S_ + qidx)) * H_ + h) * S_ + key];
        sv[kt][i] = mb ? s : MASK_FILL_F;
      }
    }

    // ---- online softmax (row reductions across the 16-lane half) ----
    float corr[8], p0[8], p1[8];
#pragma unroll
    for (int i = 0; i < 8; ++i) {
      float mx = fmaxf(sv[0][i], sv[1][i]);
#pragma unroll
      for (int off = 1; off < 16; off <<= 1)
        mx = fmaxf(mx, __shfl_xor(mx, off, 32));
      float mnew = fmaxf(m_i[i], mx);
      corr[i] = __expf(m_i[i] - mnew);
      p0[i]   = __expf(sv[0][i] - mnew);
      p1[i]   = __expf(sv[1][i] - mnew);
      float r = p0[i] + p1[i];
#pragma unroll
      for (int off = 1; off < 16; off <<= 1)
        r += __shfl_xor(r, off, 32);
      l_i[i] = l_i[i] * corr[i] + r;
      m_i[i] = mnew;
    }
#pragma unroll
    for (int t = 0; t < 4; ++t)
#pragma unroll
      for (int i = 0; i < 8; ++i) o[t][i] *= corr[i];

    // ---- transpose P (C-layout -> A-layout) through per-wave LDS tile ----
#pragma unroll
    for (int i = 0; i < 8; ++i) {
      Ps[wv][i + lhalf * 8][l16]      = (_Float16)p0[i];
      Ps[wv][i + lhalf * 8][16 + l16] = (_Float16)p1[i];
    }
    __builtin_amdgcn_wave_barrier();

    v16h pa;   // A-frag: M = query (l16), K = key (32)
#pragma unroll
    for (int i = 0; i < 8; ++i) {
      int kk = (i < 4) ? (lhalf * 8 + 2 * i) : (16 + lhalf * 8 + 2 * (i - 4));
      pa[2 * i]     = Ps[wv][l16][kk];
      pa[2 * i + 1] = Ps[wv][l16][kk + 1];
    }
#pragma unroll
    for (int dt = 0; dt < 4; ++dt) {
      v16h vb;  // B-frag: N = d (dt*16 + l16), K = key
#pragma unroll
      for (int i = 0; i < 8; ++i) {
        int kk = (i < 4) ? (lhalf * 8 + 2 * i) : (16 + lhalf * 8 + 2 * (i - 4));
        vb[2 * i]     = Vt[dt * 16 + l16][kk];
        vb[2 * i + 1] = Vt[dt * 16 + l16][kk + 1];
      }
      o[dt] = __builtin_amdgcn_wmma_f32_16x16x32_f16(
          false, pa, false, vb, (short)0, o[dt], false, false);
    }
  }

  // ---- normalize and write out[b, q, h*64 + d] ----
#pragma unroll
  for (int i = 0; i < 8; ++i) {
    float inv = 1.0f / l_i[i];
    int qidx = q0 + wv * 16 + i + lhalf * 8;
#pragma unroll
    for (int dt = 0; dt < 4; ++dt) {
      int d = dt * 16 + l16;
      out[((size_t)(b * S_ + qidx)) * DM + h * HD + d] = o[dt][i] * inv;
    }
  }
}

// ---------------------------------------------------------------------------
extern "C" void kernel_launch(void* const* d_in, const int* in_sizes, int n_in,
                              void* d_out, int out_size, void* d_ws,
                              size_t ws_size, hipStream_t stream) {
  const float*   x    = (const float*)d_in[0];
  const float*   W    = (const float*)d_in[1];
  const uint8_t* mask = (const uint8_t*)d_in[2];
  float*         out  = (float*)d_out;

  char* ws = (char*)d_ws;
  _Float16* Xh = (_Float16*)ws;                                   // 8 MB
  _Float16* Wh = (_Float16*)(ws + (size_t)ROWS * DM * 2);         // 6 MB
  _Float16* Qh = (_Float16*)(ws + (size_t)ROWS * DM * 2
                                + (size_t)NCOL * DM * 2);         // 8 MB each
  _Float16* Kh = Qh + (size_t)B_ * H_ * S_ * HD;
  _Float16* Vh = Kh + (size_t)B_ * H_ * S_ * HD;

  const int nX = ROWS * DM;
  const int nW = NCOL * DM;
  cvt_f32_f16<<<(nX + 255) / 256, 256, 0, stream>>>(x, Xh, nX);
  cvt_f32_f16<<<(nW + 255) / 256, 256, 0, stream>>>(W, Wh, nW);

  qkv_gemm<<<dim3(NCOL / 64, ROWS / 64), 128, 0, stream>>>(Xh, Wh, Qh, Kh, Vh);

  flash_attn<<<dim3(S_ / 64, B_ * H_), 128, 0, stream>>>(Qh, Kh, Vh, mask, out);
}
